// RNNLanguageModel_103079215117
// MI455X (gfx1250) — matrix operand order repaired and checked by reading, not compiled, not run on hardware
//
#include <hip/hip_runtime.h>

// Problem constants (from reference)
static constexpr int V = 32000;
static constexpr int E = 128;
static constexpr int H = 128;
static constexpr int B = 128;
static constexpr int T = 1024;

typedef __attribute__((ext_vector_type(16))) __bf16   v16bf;
typedef __attribute__((ext_vector_type(8)))  float    v8f;
typedef __attribute__((ext_vector_type(4)))  unsigned v4u;

union FragU { v16bf v; unsigned u[8]; v4u q[2]; };

// one-op truncating pack of two f32 into packed bf16 (v_perm_b32)
static __device__ __forceinline__ unsigned packtrunc_bf16x2(unsigned lo_u, unsigned hi_u) {
  return __builtin_amdgcn_perm(hi_u, lo_u, 0x07060302u);  // { hi[31:16], lo[31:16] }
}
// rounded scalar f32 -> bf16 bits (2 VALU ops)
static __device__ __forceinline__ unsigned short f2bf_rn(float f) {
  union { float f; unsigned u; } x; x.f = f;
  return (unsigned short)((x.u + 0x8000u) >> 16);
}
static __device__ __forceinline__ float dev_tanh(float v) {
#if __has_builtin(__builtin_amdgcn_tanhf)
  return __builtin_amdgcn_tanhf(v);
#elif __has_builtin(__builtin_amdgcn_tanh_f32)
  return __builtin_amdgcn_tanh_f32(v);
#else
  return tanhf(v);
#endif
}

// ---- bulk f32 -> bf16 conversion (8 elements / thread, v_perm packing) ----
__global__ void __launch_bounds__(256)
cvt_f32_bf16(const float* __restrict__ in, unsigned short* __restrict__ out, long n) {
  const long i = ((long)blockIdx.x * 256 + threadIdx.x) * 8;
  if (i + 8 <= n) {
    const v4u a = *(const v4u*)(in + i);
    const v4u b = *(const v4u*)(in + i + 4);
    v4u r;
    r.x = packtrunc_bf16x2(a.x, a.y);
    r.y = packtrunc_bf16x2(a.z, a.w);
    r.z = packtrunc_bf16x2(b.x, b.y);
    r.w = packtrunc_bf16x2(b.z, b.w);
    *(v4u*)(out + i) = r;
  }
}

// ---- WMMA fragment builders (CDNA5 16-bit layouts, ISA 7.12.2) ----
// A 16x32: lanes 0-15 -> row=lane,    K = {k0+0..7, k0+16..23}
//          lanes 16-31 -> row=lane-16, K = {k0+8..15, k0+24..31}
static __device__ __forceinline__ v16bf fragA_bf16(const unsigned short* __restrict__ rowPtr,
                                                   int k0, int lane) {
  const int khalf = (lane >> 4) << 3;
  FragU f;
  f.q[0] = *(const v4u*)(rowPtr + k0 + khalf);
  f.q[1] = *(const v4u*)(rowPtr + k0 + khalf + 16);
  return f.v;
}
static __device__ __forceinline__ v16bf fragA_lds(const unsigned short* hl,
                                                  int row, int k0, int lane) {
  const int khalf = (lane >> 4) << 3;
  const unsigned short* p = hl + row * H + k0 + khalf;
  FragU f;
  f.q[0] = *(const v4u*)(p);
  f.q[1] = *(const v4u*)(p + 16);
  return f.v;
}
// B 32x16: lane n (0-15) = col n holds K = {k0+0..7, k0+16..23}; lanes 16-31 hold K = {k0+8..15, k0+24..31}
static __device__ __forceinline__ v16bf fragB_bf16(const unsigned short* __restrict__ rowPtr,
                                                   int k0, int lane) {
  const unsigned short* p = rowPtr + k0 + ((lane >> 4) << 3);
  FragU f;
  f.q[0] = *(const v4u*)(p);
  f.q[1] = *(const v4u*)(p + 16);
  return f.v;
}

static __device__ __forceinline__ v8f wmma_bf16(v16bf a, v16bf b, v8f c) {
  return __builtin_amdgcn_wmma_f32_16x16x32_bf16(false, a, false, b, (short)0, c,
                                                 false, false);
}

// ---- pre0[t][b][:] = emb[x[b][t]] @ W_ih0^T + b_ih0 + b_hh0 ----
// grid.x = (B*T)/16 M-tiles; 8 waves cover the 8 N-tiles of H=128.
__global__ void __launch_bounds__(256)
embed_gemm(const int* __restrict__ x, const unsigned short* __restrict__ embBf,
           const unsigned short* __restrict__ Wbf, const float* __restrict__ b1,
           const float* __restrict__ b2, float* __restrict__ pre) {
  const int lane = threadIdx.x & 31;
  const int wave = threadIdx.x >> 5;
  const int l = lane & 15;
  const long r = (long)blockIdx.x * 16 + l;        // flat row = t*B + b
  const int t = (int)(r / B), bb = (int)(r % B);
  const unsigned short* arow = embBf + (size_t)x[bb * T + t] * E;
  const unsigned short* brow = Wbf + (size_t)(wave * 16 + l) * E;
  v8f acc = {};
#pragma unroll
  for (int kt = 0; kt < 4; ++kt)
    acc = wmma_bf16(fragA_bf16(arow, kt * 32, lane), fragB_bf16(brow, kt * 32, lane), acc);
  const int ncol = wave * 16 + l;
  const float bias = b1[ncol] + b2[ncol];
  const int mhalf = (lane >> 4) * 8;
#pragma unroll
  for (int i = 0; i < 8; ++i) {
    const size_t rr = (size_t)blockIdx.x * 16 + mhalf + i;
    pre[rr * H + ncol] = acc[i] + bias;
  }
}

// ---- pre1[t][b][:] = out0[t][b][:] @ W_ih1^T + b_ih1 + b_hh1 ----
__global__ void __launch_bounds__(256)
seq_gemm(const unsigned short* __restrict__ in, const unsigned short* __restrict__ Wbf,
         const float* __restrict__ b1, const float* __restrict__ b2,
         float* __restrict__ pre) {
  const int lane = threadIdx.x & 31;
  const int wave = threadIdx.x >> 5;
  const int l = lane & 15;
  const unsigned short* arow = in + ((size_t)blockIdx.x * 16 + l) * H;
  const unsigned short* brow = Wbf + (size_t)(wave * 16 + l) * H;
  v8f acc = {};
#pragma unroll
  for (int kt = 0; kt < 4; ++kt)
    acc = wmma_bf16(fragA_bf16(arow, kt * 32, lane), fragB_bf16(brow, kt * 32, lane), acc);
  const int ncol = wave * 16 + l;
  const float bias = b1[ncol] + b2[ncol];
  const int mhalf = (lane >> 4) * 8;
#pragma unroll
  for (int i = 0; i < 8; ++i) {
    const size_t rr = (size_t)blockIdx.x * 16 + mhalf + i;
    pre[rr * H + ncol] = acc[i] + bias;
  }
}

// ---- serial recurrence h_t = tanh(pre_t + h_{t-1} @ W_hh^T) ----
// Batch rows independent: 8 WGs x 16 rows. W_hh fragments pinned in registers.
// h ping-pongs between two LDS buffers -> ONE barrier per step; pre_t loads are
// software-pipelined one step ahead so global latency is off the serial chain.
__global__ void __launch_bounds__(256)
rnn_rec(const float* __restrict__ pre, const unsigned short* __restrict__ WhhBf,
        unsigned short* __restrict__ outSeq, float* __restrict__ hiddenOut) {
  __shared__ unsigned short hlds[2][16 * H];
  const int lane = threadIdx.x & 31;
  const int wave = threadIdx.x >> 5;
  const int l = lane & 15;
  const int bRow0 = blockIdx.x * 16;
  const unsigned short* brow = WhhBf + (size_t)(wave * 16 + l) * H;
  const v16bf wb0 = fragB_bf16(brow, 0, lane);
  const v16bf wb1 = fragB_bf16(brow, 32, lane);
  const v16bf wb2 = fragB_bf16(brow, 64, lane);
  const v16bf wb3 = fragB_bf16(brow, 96, lane);

  for (int i = threadIdx.x; i < 2 * 16 * H; i += 256) hlds[0][i] = 0;

  const int ncol = wave * 16 + l;
  const int mhalf = (lane >> 4) * 8;

  // prologue: pre for t=0
  float pc[8], pn[8];
  {
    const float* p0 = pre + (size_t)bRow0 * H;
#pragma unroll
    for (int i = 0; i < 8; ++i) pc[i] = p0[(size_t)(mhalf + i) * H + ncol];
  }
  __syncthreads();

  float hv[8];
#pragma unroll 1
  for (int t = 0; t < T; ++t) {
    // issue next step's pre loads; consumed only after the barrier below,
    // so their latency overlaps this step's WMMA + tanh.
    if (t + 1 < T) {
      const float* ptn = pre + ((size_t)(t + 1) * B + bRow0) * H;
#pragma unroll
      for (int i = 0; i < 8; ++i) pn[i] = ptn[(size_t)(mhalf + i) * H + ncol];
      __builtin_prefetch(ptn + (size_t)B * H + (size_t)mhalf * H + ncol, 0, 0);
    }

    const unsigned short* rb = hlds[(t + 1) & 1];  // holds h_{t-1}
    unsigned short*       wbuf = hlds[t & 1];      // receives h_t
    const v16bf a0 = fragA_lds(rb, l, 0, lane);
    const v16bf a1 = fragA_lds(rb, l, 32, lane);
    const v16bf a2 = fragA_lds(rb, l, 64, lane);
    const v16bf a3 = fragA_lds(rb, l, 96, lane);

    v8f acc = {};
    acc = wmma_bf16(a0, wb0, acc);
    acc = wmma_bf16(a1, wb1, acc);
    acc = wmma_bf16(a2, wb2, acc);
    acc = wmma_bf16(a3, wb3, acc);

    unsigned short* os = outSeq + ((size_t)t * B + bRow0) * H;
#pragma unroll
    for (int i = 0; i < 8; ++i) {
      const int m = mhalf + i;
      float v = acc[i] + pc[i];
      v = dev_tanh(v);
      hv[i] = v;
      const unsigned short bv = f2bf_rn(v);
      wbuf[m * H + ncol] = bv;
      os[(size_t)m * H + ncol] = bv;
    }
    __syncthreads();  // h_t visible; also orders next step's write into rb

#pragma unroll
    for (int i = 0; i < 8; ++i) pc[i] = pn[i];
  }
#pragma unroll
  for (int i = 0; i < 8; ++i)
    hiddenOut[(size_t)(bRow0 + mhalf + i) * H + ncol] = hv[i];
}

// ---- logits = last @ W_out^T + b_out  (M=128, N=32000, K=128) ----
__global__ void __launch_bounds__(256)
logits_gemm(const unsigned short* __restrict__ last,
            const unsigned short* __restrict__ WoutBf,
            const float* __restrict__ bout, float* __restrict__ logits) {
  const int lane = threadIdx.x & 31;
  const int wave = threadIdx.x >> 5;
  const int l = lane & 15;
  const int n0 = blockIdx.x * 16;
  const int m0 = wave * 16;
  const unsigned short* arow = last + (size_t)(m0 + l) * H;
  const unsigned short* brow = WoutBf + (size_t)(n0 + l) * H;
  v8f acc = {};
#pragma unroll
  for (int kt = 0; kt < 4; ++kt)
    acc = wmma_bf16(fragA_bf16(arow, kt * 32, lane), fragB_bf16(brow, kt * 32, lane), acc);
  const int ncol = n0 + l;
  const float bias = bout[ncol];
  const int mhalf = (lane >> 4) * 8;
#pragma unroll
  for (int i = 0; i < 8; ++i)
    logits[(size_t)(m0 + mhalf + i) * V + ncol] = acc[i] + bias;
}

static inline void launch_cvt(const float* in, unsigned short* out, long n,
                              hipStream_t s) {
  const long threads = (n + 7) / 8;
  cvt_f32_bf16<<<(unsigned)((threads + 255) / 256), 256, 0, s>>>(in, out, n);
}

extern "C" void kernel_launch(void* const* d_in, const int* in_sizes, int n_in,
                              void* d_out, int out_size, void* d_ws, size_t ws_size,
                              hipStream_t stream) {
  (void)in_sizes; (void)n_in; (void)out_size; (void)ws_size;
  const int*   x    = (const int*)d_in[0];
  const float* emb  = (const float*)d_in[1];
  const float* Wih0 = (const float*)d_in[2];
  const float* Whh0 = (const float*)d_in[3];
  const float* bih0 = (const float*)d_in[4];
  const float* bhh0 = (const float*)d_in[5];
  const float* Wih1 = (const float*)d_in[6];
  const float* Whh1 = (const float*)d_in[7];
  const float* bih1 = (const float*)d_in[8];
  const float* bhh1 = (const float*)d_in[9];
  const float* Wout = (const float*)d_in[10];
  const float* bout = (const float*)d_in[11];

  float* out    = (float*)d_out;
  float* logits = out;                           // [B][V]
  float* hidden = out + (size_t)B * V;           // [2][B][H]

  // workspace layout (all regions 256B-aligned by construction)
  char* ws = (char*)d_ws;
  float* preBuf = (float*)ws;                         ws += (size_t)T * B * H * 4;
  unsigned short* seqBuf  = (unsigned short*)ws;      ws += (size_t)T * B * H * 2;
  unsigned short* embBf   = (unsigned short*)ws;      ws += (size_t)V * E * 2;
  unsigned short* WoutBf  = (unsigned short*)ws;      ws += (size_t)V * H * 2;
  unsigned short* Wih0Bf  = (unsigned short*)ws;      ws += (size_t)H * H * 2;
  unsigned short* Whh0Bf  = (unsigned short*)ws;      ws += (size_t)H * H * 2;
  unsigned short* Wih1Bf  = (unsigned short*)ws;      ws += (size_t)H * H * 2;
  unsigned short* Whh1Bf  = (unsigned short*)ws;      // end

  // one-time f32 -> bf16 of all matrix operands
  launch_cvt(emb,  embBf,  (long)V * E, stream);
  launch_cvt(Wout, WoutBf, (long)V * H, stream);
  launch_cvt(Wih0, Wih0Bf, (long)H * H, stream);
  launch_cvt(Whh0, Whh0Bf, (long)H * H, stream);
  launch_cvt(Wih1, Wih1Bf, (long)H * H, stream);
  launch_cvt(Whh1, Whh1Bf, (long)H * H, stream);

  embed_gemm<<<(B * T) / 16, 256, 0, stream>>>(x, embBf, Wih0Bf, bih0, bhh0, preBuf);
  rnn_rec<<<B / 16, 256, 0, stream>>>(preBuf, Whh0Bf, seqBuf, hidden);
  seq_gemm<<<(B * T) / 16, 256, 0, stream>>>(seqBuf, Wih1Bf, bih1, bhh1, preBuf);
  rnn_rec<<<B / 16, 256, 0, stream>>>(preBuf, Whh1Bf, seqBuf, hidden + (size_t)B * H);
  logits_gemm<<<V / 16, 256, 0, stream>>>(seqBuf + (size_t)(T - 1) * B * H,
                                          WoutBf, bout, logits);
}
